// QuantizedLinear_48318382080341
// MI455X (gfx1250) — compile-verified, matrix operand order checked by reading, not run on hardware
//
#include <hip/hip_runtime.h>

// ---------------------------------------------------------------------------
// QuantizedLinear: out[M,N] = X[M,K] @ W[N,K]^T * scale + bias
//   M = 16384, N = 4096, K = 4096; W int8-in-int32, zp = 0.
//
// Fast path (needs 160MB workspace):
//   1) pack kernels: W -> bf16 (32MB), X -> bf16 (128MB) in d_ws
//   2) GEMM: TDM tensor_load_to_lds double-buffered tiles (padded rows),
//      V_WMMA_F32_16X16X32_BF16, 64x64 wave tiles (1:1 ds:wmma ratio).
// Fallback: fused convert-in-loop kernel.
// ---------------------------------------------------------------------------

typedef __attribute__((ext_vector_type(16))) __bf16 v16bf;
typedef __attribute__((ext_vector_type(8)))  __bf16 v8bf;
typedef __attribute__((ext_vector_type(8)))  float  v8f;
typedef __attribute__((ext_vector_type(4)))  float  f4;
typedef __attribute__((ext_vector_type(4)))  int    i4;
typedef __attribute__((ext_vector_type(4)))  unsigned int u32x4;
typedef __attribute__((ext_vector_type(8)))  int    i32x8;
typedef __attribute__((ext_vector_type(4)))  int    i32x4;

// ========================== prepack kernels ================================

__global__ __launch_bounds__(256)
void pack_w_kernel(const int* __restrict__ q, const int* __restrict__ zp_p,
                   __bf16* __restrict__ out, long long total)
{
    long long i = ((long long)blockIdx.x * blockDim.x + threadIdx.x) * 8;
    if (i >= total) return;
    const int zp = *zp_p;
    i4 a = *(const i4*)(q + i);
    i4 b = *(const i4*)(q + i + 4);
    v8bf o;
#pragma unroll
    for (int j = 0; j < 4; ++j) {
        o[j]     = (__bf16)(float)(a[j] - zp);   // exact in bf16 (|v| <= 256)
        o[j + 4] = (__bf16)(float)(b[j] - zp);
    }
    *(v8bf*)(out + i) = o;
}

__global__ __launch_bounds__(256)
void pack_x_kernel(const float* __restrict__ x, __bf16* __restrict__ out,
                   long long total)
{
    long long i = ((long long)blockIdx.x * blockDim.x + threadIdx.x) * 8;
    if (i >= total) return;
    f4 a = *(const f4*)(x + i);
    f4 b = *(const f4*)(x + i + 4);
    v8bf o;
#pragma unroll
    for (int j = 0; j < 4; ++j) {
        o[j]     = (__bf16)a[j];
        o[j + 4] = (__bf16)b[j];
    }
    *(v8bf*)(out + i) = o;
}

// ========================== TDM descriptor helper ==========================
// 2D tile load: tile_dim0 = 64 elems (128B rows), tile_dim1 = nrows,
// data_size = 2B, LDS padding: +4 DWORDs (16B) every 32 DWORDs (128B)
// -> LDS row stride 144B = 36 banks (conflict-free b128 reads).

__device__ __forceinline__ void tdm_load_tile(unsigned lds_addr,
                                              unsigned long long gaddr,
                                              unsigned tensor_d0,
                                              unsigned tensor_d1,
                                              unsigned long long d0_stride,
                                              unsigned nrows)
{
    u32x4 g0;
    g0[0] = 1u;                                            // count=1 (valid)
    g0[1] = lds_addr;                                      // LDS byte address
    g0[2] = (unsigned)(gaddr & 0xffffffffull);             // global_addr lo
    g0[3] = (unsigned)((gaddr >> 32) & 0x1ffffffull)       // global_addr hi
          | 0x80000000u;                                   // type=2 ("image")

    i32x8 g1;
    g1[0] = (1 << 16)      // data_size = 1 -> 2 bytes
          | (1 << 20)      // pad_enable
          | (4 << 22)      // pad_interval: 32 DWORDs (=128B = tile row)
          | (3 << 25);     // pad_amount: 4 DWORDs (=16B)
    g1[1] = (int)((tensor_d0 & 0xffffu) << 16);            // dim0[15:0]
    g1[2] = (int)(((tensor_d0 >> 16) & 0xffffu)            // dim0[31:16]
          | ((tensor_d1 & 0xffffu) << 16));                // dim1[15:0]
    g1[3] = (int)(((tensor_d1 >> 16) & 0xffffu)            // dim1[31:16]
          | (64u << 16));                                  // tile_dim0 = 64
    g1[4] = (int)(nrows & 0xffffu);                        // tile_dim1 = rows
    g1[5] = (int)(d0_stride & 0xffffffffull);              // dim0_stride lo
    g1[6] = (int)((d0_stride >> 32) & 0xffffull);          // dim0_stride hi
    g1[7] = 0;

    i32x4 z4 = {0, 0, 0, 0};
#if __clang_major__ >= 23
    i32x8 z8 = {0, 0, 0, 0, 0, 0, 0, 0};
    __builtin_amdgcn_tensor_load_to_lds(g0, g1, z4, z4, z8, 0);
#else
    __builtin_amdgcn_tensor_load_to_lds(g0, g1, z4, z4, 0);
#endif
}

// ========================== bf16 TDM GEMM ==================================
// Block tile 128(M) x 256(N), 8 waves in 2x4, each wave owns 64x64
// (4x4 WMMA tiles, 128 accumulator VGPRs, 4x reuse on A and B).

#define GBM 128
#define GBN 256
#define GBK 64
#define GLDSK 72                        // 64 + 8 pad bf16 elems = 144B rows
#define GA_ELEMS (GBM * GLDSK)          // A tile:  9216 elems
#define GW_ELEMS (GBN * GLDSK)          // W tile: 18432 elems
#define GBUF_ELEMS (GA_ELEMS + GW_ELEMS)

__global__ __launch_bounds__(256, 1)
void gemm_bf16_tdm(const __bf16* __restrict__ Xb,
                   const __bf16* __restrict__ Wb,
                   const float*  __restrict__ scale_p,
                   const float*  __restrict__ bias,
                   float*        __restrict__ out,
                   int M, int N, int K)
{
    __shared__ __align__(16) __bf16 smem[2 * GBUF_ELEMS];  // ~108KB of 320KB

    const int tid  = threadIdx.x;
    const int lane = tid & 31;
    const int wave = tid >> 5;
    const int m0 = blockIdx.y * GBM;
    const int n0 = blockIdx.x * GBN;

    const int wm = (wave >> 2) * 64;         // 0,64
    const int wn = (wave &  3) * 64;         // 0,64,128,192
    const int lh = lane & 15;
    const int kh = (lane >> 4) * 8;          // K sub-offset (A/B), +8 rows (D)

    const unsigned lds0 = (unsigned)(unsigned long long)(const void*)&smem[0];
    const unsigned long long xg0 = (unsigned long long)Xb + (unsigned long long)m0 * K * 2;
    const unsigned long long wg0 = (unsigned long long)Wb + (unsigned long long)n0 * K * 2;

    v8f acc[4][4] = {};

    // prologue: wave 0 DMAs tile 0 into buffer 0 via the Tensor Data Mover
    if (wave == 0) {
        tdm_load_tile(lds0,                   xg0, K, M, (unsigned long long)K, GBM);
        tdm_load_tile(lds0 + GA_ELEMS * 2,    wg0, K, N, (unsigned long long)K, GBN);
        __builtin_amdgcn_s_wait_tensorcnt(0);
    }
    __syncthreads();

    const int NT = K / GBK;                  // 64 iterations
    for (int kt = 0; kt < NT; ++kt) {
        const int buf = kt & 1;

        // issue next tile's DMA into the other buffer (consumed last iter)
        if (wave == 0 && kt + 1 < NT) {
            const unsigned long long koff = (unsigned long long)(kt + 1) * GBK * 2;
            const unsigned dst = lds0 + (buf ^ 1) * GBUF_ELEMS * 2;
            tdm_load_tile(dst,                 xg0 + koff, K, M, (unsigned long long)K, GBM);
            tdm_load_tile(dst + GA_ELEMS * 2,  wg0 + koff, K, N, (unsigned long long)K, GBN);
        }

        const __bf16* ab = smem + buf * GBUF_ELEMS;
        const __bf16* wb = ab + GA_ELEMS;

#pragma unroll
        for (int ks = 0; ks < 2; ++ks) {     // two K=32 sub-steps of the 64-tile
            const int k32 = ks * 32;
            v16bf afr[4], bfr[4];
#pragma unroll
            for (int mi = 0; mi < 4; ++mi) {
                const __bf16* p = ab + (wm + mi * 16 + lh) * GLDSK + k32 + kh;
                v8bf lo = *(const v8bf*)p;
                v8bf hi = *(const v8bf*)(p + 16);
                afr[mi] = __builtin_shufflevector(lo, hi,
                    0,1,2,3,4,5,6,7,8,9,10,11,12,13,14,15);
            }
#pragma unroll
            for (int ni = 0; ni < 4; ++ni) {
                const __bf16* p = wb + (wn + ni * 16 + lh) * GLDSK + k32 + kh;
                v8bf lo = *(const v8bf*)p;
                v8bf hi = *(const v8bf*)(p + 16);
                bfr[ni] = __builtin_shufflevector(lo, hi,
                    0,1,2,3,4,5,6,7,8,9,10,11,12,13,14,15);
            }
#pragma unroll
            for (int mi = 0; mi < 4; ++mi)
#pragma unroll
                for (int ni = 0; ni < 4; ++ni)
                    acc[mi][ni] = __builtin_amdgcn_wmma_f32_16x16x32_bf16(
                        false, afr[mi], false, bfr[ni],
                        (short)0, acc[mi][ni], false, false);
        }

        if (wave == 0 && kt + 1 < NT)
            __builtin_amdgcn_s_wait_tensorcnt(0);   // next buffer landed
        __syncthreads();
    }

    // epilogue: D layout VGPR r -> row r (+8 for upper half-wave)
    const float scale = *scale_p;
#pragma unroll
    for (int ni = 0; ni < 4; ++ni) {
        const int col = n0 + wn + ni * 16 + lh;
        const float bv = bias[col];
#pragma unroll
        for (int mi = 0; mi < 4; ++mi) {
            const int rbase = m0 + wm + mi * 16 + kh;
            v8f c = acc[mi][ni];
#pragma unroll
            for (int r = 0; r < 8; ++r)
                __builtin_nontemporal_store(c[r] * scale + bv,
                    out + (size_t)(rbase + r) * N + col);
        }
    }
}

// ========================== fused fallback =================================

#define BM 128
#define BN 128
#define BK 32
#define LDSK 40
#define ABUF_ELEMS (BM * LDSK)
#define BUF_ELEMS  (2 * ABUF_ELEMS)

__global__ __launch_bounds__(256)
void qlinear_fused_fallback(const float* __restrict__ X,
                            const int*   __restrict__ Wq,
                            const float* __restrict__ scale_p,
                            const int*   __restrict__ zp_p,
                            const float* __restrict__ bias,
                            float*       __restrict__ out,
                            int M, int N, int K)
{
    __shared__ __align__(16) __bf16 smem[2 * BUF_ELEMS];

    const int tid  = threadIdx.x;
    const int lane = tid & 31;
    const int wave = tid >> 5;
    const int m0 = blockIdx.y * BM;
    const int n0 = blockIdx.x * BN;

    const int lrow = tid >> 1;
    const int lk   = (tid & 1) * 16;
    const float* xsrc = X  + (size_t)(m0 + lrow) * K + lk;
    const int*   wsrc = Wq + (size_t)(n0 + lrow) * K + lk;
    const int zp = *zp_p;

    f4 xr[4];
    i4 wr[4];

    auto load_tile = [&](int kt) {
        const f4* xv = (const f4*)(xsrc + (size_t)kt * BK);
        const i4* wv = (const i4*)(wsrc + (size_t)kt * BK);
#pragma unroll
        for (int i = 0; i < 4; ++i) { xr[i] = xv[i]; wr[i] = wv[i]; }
    };

    auto store_tile = [&](int buf) {
        __bf16* abase = smem + buf * BUF_ELEMS;
        __bf16* wbase = abase + ABUF_ELEMS;
        v8bf xa, xb, wa, wb;
        const float* xf = (const float*)xr;
        const int*   wi = (const int*)wr;
#pragma unroll
        for (int i = 0; i < 8; ++i) {
            xa[i] = (__bf16)xf[i];
            xb[i] = (__bf16)xf[i + 8];
            wa[i] = (__bf16)(float)(wi[i]     - zp);
            wb[i] = (__bf16)(float)(wi[i + 8] - zp);
        }
        *(v8bf*)(abase + lrow * LDSK + lk)     = xa;
        *(v8bf*)(abase + lrow * LDSK + lk + 8) = xb;
        *(v8bf*)(wbase + lrow * LDSK + lk)     = wa;
        *(v8bf*)(wbase + lrow * LDSK + lk + 8) = wb;
    };

    const int wm = (wave >> 1) * 32;
    const int wn = (wave &  1) * 64;
    const int lh = lane & 15;
    const int kh = (lane >> 4) * 8;

    v8f acc[2][4] = {};

    load_tile(0);
    store_tile(0);
    __syncthreads();

    const int NT = K / BK;
    for (int kt = 0; kt < NT; ++kt) {
        const int buf = kt & 1;
        if (kt + 1 < NT) load_tile(kt + 1);

        const __bf16* abase = smem + buf * BUF_ELEMS;
        const __bf16* wbase = abase + ABUF_ELEMS;

        v16bf afrag[2], bfrag[4];
#pragma unroll
        for (int mi = 0; mi < 2; ++mi) {
            const __bf16* p = abase + (wm + mi * 16 + lh) * LDSK + kh;
            v8bf lo = *(const v8bf*)p;
            v8bf hi = *(const v8bf*)(p + 16);
            afrag[mi] = __builtin_shufflevector(lo, hi,
                0,1,2,3,4,5,6,7,8,9,10,11,12,13,14,15);
        }
#pragma unroll
        for (int ni = 0; ni < 4; ++ni) {
            const __bf16* p = wbase + (wn + ni * 16 + lh) * LDSK + kh;
            v8bf lo = *(const v8bf*)p;
            v8bf hi = *(const v8bf*)(p + 16);
            bfrag[ni] = __builtin_shufflevector(lo, hi,
                0,1,2,3,4,5,6,7,8,9,10,11,12,13,14,15);
        }

#pragma unroll
        for (int mi = 0; mi < 2; ++mi)
#pragma unroll
            for (int ni = 0; ni < 4; ++ni)
                acc[mi][ni] = __builtin_amdgcn_wmma_f32_16x16x32_bf16(
                    false, afrag[mi], false, bfrag[ni],
                    (short)0, acc[mi][ni], false, false);

        if (kt + 1 < NT) store_tile((kt + 1) & 1);
        __syncthreads();
    }

    const float scale = *scale_p;
#pragma unroll
    for (int ni = 0; ni < 4; ++ni) {
        const int col = n0 + wn + ni * 16 + lh;
        const float bv = bias[col];
#pragma unroll
        for (int mi = 0; mi < 2; ++mi) {
            const int rbase = m0 + wm + mi * 16 + kh;
            v8f c = acc[mi][ni];
#pragma unroll
            for (int r = 0; r < 8; ++r)
                out[(size_t)(rbase + r) * N + col] = c[r] * scale + bv;
        }
    }
}

// ========================== host launcher ==================================

extern "C" void kernel_launch(void* const* d_in, const int* in_sizes, int n_in,
                              void* d_out, int out_size, void* d_ws, size_t ws_size,
                              hipStream_t stream) {
    const float* x  = (const float*)d_in[0];
    const int*   wq = (const int*)  d_in[1];
    const float* sc = (const float*)d_in[2];
    const int*   zp = (const int*)  d_in[3];
    const float* b  = (const float*)d_in[4];
    float* out = (float*)d_out;

    const int N = in_sizes[4];          // 4096
    const int K = in_sizes[1] / N;      // 4096
    const int M = in_sizes[0] / K;      // 16384

    const size_t wbytes = (size_t)N * K * sizeof(__bf16);   // 32MB
    const size_t xbytes = (size_t)M * K * sizeof(__bf16);   // 128MB

    if (ws_size >= wbytes + xbytes) {
        __bf16* Wb = (__bf16*)d_ws;
        __bf16* Xb = (__bf16*)((char*)d_ws + wbytes);

        const long long wtotal = (long long)N * K;
        const long long xtotal = (long long)M * K;
        pack_w_kernel<<<(int)(wtotal / (256 * 8)), 256, 0, stream>>>(wq, zp, Wb, wtotal);
        pack_x_kernel<<<(int)(xtotal / (256 * 8)), 256, 0, stream>>>(x, Xb, xtotal);

        dim3 grid(N / GBN, M / GBM);    // (16, 128)
        gemm_bf16_tdm<<<grid, 256, 0, stream>>>(Xb, Wb, sc, b, out, M, N, K);
    } else {
        dim3 grid(N / BN, M / BM);
        qlinear_fused_fallback<<<grid, 256, 0, stream>>>(x, wq, sc, zp, b, out, M, N, K);
    }
}
